// MHA_4956392260105
// MI455X (gfx1250) — matrix operand order repaired and checked by reading, not compiled, not run on hardware
//
#include <hip/hip_runtime.h>

// ---------------------------------------------------------------------------
// CDNA5 (gfx1250) fused MHA: bf16 WMMA GEMMs (async-to-LDS double buffered)
// + flash attention.  B=2, T=2048, DIM=1024, NH=16, HD=64.
// ---------------------------------------------------------------------------

typedef __attribute__((ext_vector_type(16))) __bf16 bf16x16;
typedef __attribute__((ext_vector_type(8)))  __bf16 bf16x8;
typedef __attribute__((ext_vector_type(8)))  float  f32x8;
typedef __attribute__((ext_vector_type(4)))  float  f32x4;
typedef __attribute__((ext_vector_type(4)))  unsigned int u32x4;

union BF16x16 { bf16x16 v; bf16x8 h[2]; u32x4 q[2]; };
union BF16x8U { bf16x8 v; u32x4 q; };

static __device__ __forceinline__ f32x8 wmma_bf16(const BF16x16& a,
                                                  const BF16x16& b,
                                                  f32x8 c) {
  // 8 args: (neg_a, A, neg_b, B, c_mod, C, reuse_a, reuse_b)
  return __builtin_amdgcn_wmma_f32_16x16x32_bf16(false, a.v, false, b.v,
                                                 (short)0, c, false, false);
}

// Async DMA: copy 16B from global to LDS, tracked by ASYNCcnt (CDNA5 §08).
static __device__ __forceinline__ void async_copy_b128(unsigned lds_off,
                                                       const void* gptr) {
  asm volatile("global_load_async_to_lds_b128 %0, %1, off"
               :: "v"(lds_off), "v"((unsigned long long)(size_t)gptr)
               : "memory");
}

// ---------------------------------------------------------------------------
// Elementwise fp32 -> bf16 (8 elements / thread, b128 in / b128 out)
// ---------------------------------------------------------------------------
__global__ __launch_bounds__(256) void cvt_bf16(const float* __restrict__ s,
                                                __bf16* __restrict__ d,
                                                int n8) {
  int i = blockIdx.x * blockDim.x + threadIdx.x;
  if (i >= n8) return;
  const f32x4* sp = (const f32x4*)s + 2 * (size_t)i;
  f32x4 a = sp[0], b = sp[1];
  BF16x8U pk;
#pragma unroll
  for (int j = 0; j < 4; ++j) pk.v[j]     = (__bf16)a[j];
#pragma unroll
  for (int j = 0; j < 4; ++j) pk.v[4 + j] = (__bf16)b[j];
  *((u32x4*)d + (size_t)i) = pk.q;
}

// ---------------------------------------------------------------------------
// Tiled transpose + convert: W[K,N] fp32 -> Wt[N,K] bf16 (32x32 LDS tiles)
// ---------------------------------------------------------------------------
__global__ __launch_bounds__(256) void transpose_cvt_bf16(
    const float* __restrict__ W, __bf16* __restrict__ Wt, int K, int N) {
  __shared__ float tile[32][33];
  const int k0 = blockIdx.y * 32, n0 = blockIdx.x * 32;
  const int tx = threadIdx.x, ty = threadIdx.y;  // (32, 8)
#pragma unroll
  for (int i = 0; i < 32; i += 8)
    tile[ty + i][tx] = W[(size_t)(k0 + ty + i) * N + n0 + tx];
  __syncthreads();
#pragma unroll
  for (int i = 0; i < 32; i += 8)
    Wt[(size_t)(n0 + ty + i) * K + k0 + tx] = (__bf16)tile[tx][ty + i];
}

// ---------------------------------------------------------------------------
// bf16 GEMM: C[M,N] (fp32) = A[M,K] @ Bt[N,K]^T, both operands bf16 row-major
// (Bt is pre-transposed so B-fragments are K-contiguous).
// WG tile 128x128, K-step 32, 8 waves, wave tile 32x64 (8 accumulators).
// Tiles staged via GLOBAL_LOAD_ASYNC_TO_LDS_B128, double buffered.
// ---------------------------------------------------------------------------
#define LDSS 40   // bf16 elements per LDS row (32 + 8 pad; 80B keeps 16B align)

__global__ __launch_bounds__(256) void gemm_bf16_async(
    const __bf16* __restrict__ A, const __bf16* __restrict__ Bt,
    float* __restrict__ C, int M, int N, int K) {
  __shared__ __bf16 Als[2][128][LDSS];
  __shared__ __bf16 Bls[2][128][LDSS];

  const int tid  = threadIdx.x;
  const int lane = tid & 31;
  const int wave = tid >> 5;
  const int wm   = wave >> 1;   // 0..3 -> m offset wm*32
  const int wn   = wave & 1;    // 0..1 -> n offset wn*64
  const int m0   = blockIdx.y * 128;
  const int n0   = blockIdx.x * 128;

  const int lh   = lane >> 4;
  const int ln   = lane & 15;
  const int kb8  = lh * 8;
  const int kb16 = lh * 16;

  // cooperative DMA mapping: thread copies 32B of A row + 32B of Bt row
  const int crow = tid >> 1;         // 0..127
  const int cseg = (tid & 1) * 16;   // bf16 element offset 0 / 16

  const int nk = K / 32;

  // prologue: issue tile 0 into buffer 0
  {
    const __bf16* ga = A  + (size_t)(m0 + crow) * K + cseg;
    const __bf16* gb = Bt + (size_t)(n0 + crow) * K + cseg;
    unsigned la = (unsigned)(size_t)&Als[0][crow][cseg];
    unsigned lb = (unsigned)(size_t)&Bls[0][crow][cseg];
    async_copy_b128(la,      ga);
    async_copy_b128(la + 16, ga + 8);
    async_copy_b128(lb,      gb);
    async_copy_b128(lb + 16, gb + 8);
  }

  f32x8 acc[2][4];
#pragma unroll
  for (int s = 0; s < 2; ++s)
#pragma unroll
    for (int j = 0; j < 4; ++j) acc[s][j] = (f32x8){};

  for (int i = 0; i < nk; ++i) {
    const int cur = i & 1;
    if (i + 1 < nk) {  // issue next tile into other buffer (overlaps compute)
      const int nxt = cur ^ 1;
      const int k0  = (i + 1) * 32;
      const __bf16* ga = A  + (size_t)(m0 + crow) * K + k0 + cseg;
      const __bf16* gb = Bt + (size_t)(n0 + crow) * K + k0 + cseg;
      unsigned la = (unsigned)(size_t)&Als[nxt][crow][cseg];
      unsigned lb = (unsigned)(size_t)&Bls[nxt][crow][cseg];
      async_copy_b128(la,      ga);
      async_copy_b128(la + 16, ga + 8);
      async_copy_b128(lb,      gb);
      async_copy_b128(lb + 16, gb + 8);
      asm volatile("s_wait_asynccnt 0x4" ::: "memory");  // tile i landed
    } else {
      asm volatile("s_wait_asynccnt 0x0" ::: "memory");
    }
    __syncthreads();

    // A fragments (2 m-subtiles): lane ln = row, chunks [kb8,+8) & [16+kb8,+8)
    BF16x16 a[2];
#pragma unroll
    for (int s = 0; s < 2; ++s) {
      const __bf16* ap = &Als[cur][wm * 32 + s * 16 + ln][0];
      a[s].q[0] = *(const u32x4*)(ap + kb8);
      a[s].q[1] = *(const u32x4*)(ap + 16 + kb8);
    }
    // B fragments (4 n-subtiles): lane ln = column, chunks [kb8,+8) & [16+kb8,+8)
#pragma unroll
    for (int j = 0; j < 4; ++j) {
      BF16x16 b;
      const __bf16* bp = &Bls[cur][wn * 64 + j * 16 + ln][0];
      b.q[0] = *(const u32x4*)(bp + kb8);
      b.q[1] = *(const u32x4*)(bp + 16 + kb8);
      acc[0][j] = wmma_bf16(a[0], b, acc[0][j]);
      acc[1][j] = wmma_bf16(a[1], b, acc[1][j]);
    }
    __syncthreads();  // compute done before buffer `cur` is overwritten
  }

  // C/D layout: lane ln = col, VGPR v = row (v + 8*lh)
#pragma unroll
  for (int s = 0; s < 2; ++s) {
    const int cm = m0 + wm * 32 + s * 16 + lh * 8;
#pragma unroll
    for (int j = 0; j < 4; ++j) {
      const int cn = n0 + wn * 64 + j * 16 + ln;
#pragma unroll
      for (int v = 0; v < 8; ++v)
        C[(size_t)(cm + v) * N + cn] = acc[s][j][v];
    }
  }
}

// ---------------------------------------------------------------------------
// RoPE on q,k + reshape to [B,NH,T,HD] bf16 (q,k) and [B,NH,HD,T] bf16 (v^T).
// ---------------------------------------------------------------------------
__global__ __launch_bounds__(256) void rope_split(
    const float* __restrict__ qkv, const float* __restrict__ sinp,
    const float* __restrict__ cosp, __bf16* __restrict__ qd,
    __bf16* __restrict__ kd, __bf16* __restrict__ vtd,
    int Bsz, int T, int NHc, int HDc) {
  int idx = blockIdx.x * blockDim.x + threadIdx.x;
  int half = HDc >> 1;
  int total = Bsz * T * NHc * half;
  if (idx >= total) return;
  int i = idx % half; int tmp = idx / half;
  int h = tmp % NHc;  tmp /= NHc;
  int t = tmp % T;    int b = tmp / T;
  int d0 = 2 * i, d1 = 2 * i + 1;
  const float* row = qkv + (size_t)(b * T + t) * (3 * NHc * HDc);
  int col = h * HDc;
  float q0 = row[col + d0],                 q1 = row[col + d1];
  float k0 = row[NHc * HDc + col + d0],     k1 = row[NHc * HDc + col + d1];
  float v0 = row[2 * NHc * HDc + col + d0], v1 = row[2 * NHc * HDc + col + d1];
  float c0 = cosp[t * HDc + d0], s0 = sinp[t * HDc + d0];
  float c1 = cosp[t * HDc + d1], s1 = sinp[t * HDc + d1];
  float qo0 = q0 * c0 - q1 * s0;
  float qo1 = q1 * c1 + q0 * s1;
  float ko0 = k0 * c0 - k1 * s0;
  float ko1 = k1 * c1 + k0 * s1;
  size_t base = ((size_t)(b * NHc + h) * T + t) * HDc;
  qd[base + d0] = (__bf16)qo0; qd[base + d1] = (__bf16)qo1;
  kd[base + d0] = (__bf16)ko0; kd[base + d1] = (__bf16)ko1;
  size_t vbase = (size_t)(b * NHc + h) * HDc;
  vtd[(vbase + d0) * T + t] = (__bf16)v0;
  vtd[(vbase + d1) * T + t] = (__bf16)v1;
}

// ---------------------------------------------------------------------------
// Flash attention (causal, online softmax). 8 waves/WG, 16 q-rows per wave,
// key blocks of 32. 8 WMMAs per key block per wave. Output bf16.
// q,k: [BH, T, 64] bf16.  vt: [BH, 64, T] bf16.  y: [B, T, NH*HD] bf16.
// ---------------------------------------------------------------------------
#define ATT_WAVES 8
#define ATT_QTILE 128
#define PSTR 40

__global__ __launch_bounds__(256) void flash_attn(
    const __bf16* __restrict__ qm, const __bf16* __restrict__ km,
    const __bf16* __restrict__ vt, __bf16* __restrict__ y, int T) {
  __shared__ __bf16 Pls[ATT_WAVES][16][PSTR];  // wave-private P transpose bounce

  const int lane = threadIdx.x & 31;
  const int wave = threadIdx.x >> 5;
  const int bh   = blockIdx.y;
  const int qb   = blockIdx.x * ATT_QTILE + wave * 16;
  const size_t qk_base = (size_t)bh * T * 64;
  const size_t vt_base = (size_t)bh * 64 * T;

  const int lh   = lane >> 4;
  const int ln   = lane & 15;
  const int kb8  = lh * 8;
  const int kb16 = lh * 16;

  BF16x16 aq0, aq1;
  {
    const __bf16* qrow = qm + qk_base + (size_t)(qb + ln) * 64;
    aq0.q[0] = *(const u32x4*)(qrow + kb8);
    aq0.q[1] = *(const u32x4*)(qrow + 16 + kb8);
    aq1.q[0] = *(const u32x4*)(qrow + 32 + kb8);
    aq1.q[1] = *(const u32x4*)(qrow + 48 + kb8);
  }

  f32x8 yacc[4] = {{}, {}, {}, {}};
  float mstate[8], lstate[8];
#pragma unroll
  for (int v = 0; v < 8; ++v) { mstate[v] = -1e30f; lstate[v] = 0.f; }

  const float scale = 0.125f;  // 1/sqrt(64)
  const int row_hi = qb + 15;

  for (int kb = 0; kb <= row_hi; kb += 32) {
    f32x8 s0 = {}, s1 = {};
    {
      const __bf16* krA = km + qk_base + (size_t)(kb + ln) * 64;
      const __bf16* krB = km + qk_base + (size_t)(kb + 16 + ln) * 64;
      BF16x16 b;
      b.q[0] = *(const u32x4*)(krA + kb8);
      b.q[1] = *(const u32x4*)(krA + 16 + kb8);
      s0 = wmma_bf16(aq0, b, s0);
      b.q[0] = *(const u32x4*)(krA + 32 + kb8);
      b.q[1] = *(const u32x4*)(krA + 48 + kb8);
      s0 = wmma_bf16(aq1, b, s0);
      b.q[0] = *(const u32x4*)(krB + kb8);
      b.q[1] = *(const u32x4*)(krB + 16 + kb8);
      s1 = wmma_bf16(aq0, b, s1);
      b.q[0] = *(const u32x4*)(krB + 32 + kb8);
      b.q[1] = *(const u32x4*)(krB + 48 + kb8);
      s1 = wmma_bf16(aq1, b, s1);
    }

    float p0a[8], p1a[8], alpha[8];
#pragma unroll
    for (int v = 0; v < 8; ++v) {
      const int rowm = qb + v + lh * 8;
      float sv0 = s0[v] * scale; if (kb + ln > rowm)      sv0 = -1e30f;
      float sv1 = s1[v] * scale; if (kb + 16 + ln > rowm) sv1 = -1e30f;
      float mx = fmaxf(sv0, sv1);
#pragma unroll
      for (int off = 1; off < 16; off <<= 1)
        mx = fmaxf(mx, __shfl_xor(mx, off, 32));
      mx = fmaxf(mx, mstate[v]);
      float a_ = __expf(mstate[v] - mx);
      float p0 = __expf(sv0 - mx);
      float p1 = __expf(sv1 - mx);
      float rs = p0 + p1;
#pragma unroll
      for (int off = 1; off < 16; off <<= 1)
        rs += __shfl_xor(rs, off, 32);
      mstate[v] = mx;
      lstate[v] = lstate[v] * a_ + rs;
      alpha[v] = a_; p0a[v] = p0; p1a[v] = p1;
    }

#pragma unroll
    for (int v = 0; v < 8; ++v) {
      const int r = v + lh * 8;
      Pls[wave][r][ln]      = (__bf16)p0a[v];
      Pls[wave][r][16 + ln] = (__bf16)p1a[v];
    }
    asm volatile("s_wait_dscnt 0" ::: "memory");  // LDS RAW within wave
    BF16x16 aP;
    const __bf16* pr = &Pls[wave][ln][0];
    aP.q[0] = *(const u32x4*)(pr + kb8);
    aP.q[1] = *(const u32x4*)(pr + 16 + kb8);

#pragma unroll
    for (int j = 0; j < 4; ++j)
#pragma unroll
      for (int v = 0; v < 8; ++v) yacc[j][v] *= alpha[v];
#pragma unroll
    for (int j = 0; j < 4; ++j) {
      BF16x16 bv;
      const __bf16* vr = vt + vt_base + (size_t)(j * 16 + ln) * T + kb + kb8;
      bv.q[0] = *(const u32x4*)(vr);
      bv.q[1] = *(const u32x4*)(vr + 16);
      yacc[j] = wmma_bf16(aP, bv, yacc[j]);
    }
  }

  const int b = bh >> 4, h = bh & 15;
#pragma unroll
  for (int j = 0; j < 4; ++j) {
    const int d = j * 16 + ln;
#pragma unroll
    for (int v = 0; v < 8; ++v) {
      const int t = qb + lh * 8 + v;
      y[(size_t)(b * T + t) * 1024 + h * 64 + d] =
          (__bf16)(yacc[j][v] / lstate[v]);
    }
  }
}

// ---------------------------------------------------------------------------
// Host-side launcher
// ---------------------------------------------------------------------------
extern "C" void kernel_launch(void* const* d_in, const int* in_sizes, int n_in,
                              void* d_out, int out_size, void* d_ws,
                              size_t ws_size, hipStream_t stream) {
  (void)in_sizes; (void)n_in; (void)out_size; (void)ws_size;
  const float* x     = (const float*)d_in[0];
  const float* sinp  = (const float*)d_in[1];
  const float* cosp  = (const float*)d_in[2];
  const float* Wqkv  = (const float*)d_in[3];
  const float* Wproj = (const float*)d_in[4];
  float* out = (float*)d_out;

  const int Bc = 2, T = 2048, DIM = 1024, NHc = 16, HDc = 64;
  const int M = Bc * T;  // 4096

  char* ws = (char*)d_ws;
  size_t off = 0;
  float* qkv = (float*)(ws + off);   off += (size_t)M * 3 * DIM * sizeof(float);
  __bf16* xbf    = (__bf16*)(ws + off); off += (size_t)M * DIM * 2;
  __bf16* wqkvT  = (__bf16*)(ws + off); off += (size_t)3 * DIM * DIM * 2;
  __bf16* wprojT = (__bf16*)(ws + off); off += (size_t)DIM * DIM * 2;
  __bf16* qbf = (__bf16*)(ws + off); off += (size_t)M * DIM * 2;
  __bf16* kbf = (__bf16*)(ws + off); off += (size_t)M * DIM * 2;
  __bf16* vtb = (__bf16*)(ws + off); off += (size_t)M * DIM * 2;
  __bf16* ybf = (__bf16*)(ws + off);

  // 0) one-time precision/layout prep (bf16 activations, transposed weights)
  int n8 = M * DIM / 8;
  cvt_bf16<<<(n8 + 255) / 256, 256, 0, stream>>>(x, xbf, n8);
  dim3 tb(32, 8);
  transpose_cvt_bf16<<<dim3(3 * DIM / 32, DIM / 32), tb, 0, stream>>>(
      Wqkv, wqkvT, DIM, 3 * DIM);
  transpose_cvt_bf16<<<dim3(DIM / 32, DIM / 32), tb, 0, stream>>>(
      Wproj, wprojT, DIM, DIM);

  // 1) qkv = x @ W_qkv   (4096 x 3072 x 1024)
  gemm_bf16_async<<<dim3(3 * DIM / 128, M / 128), 256, 0, stream>>>(
      xbf, wqkvT, qkv, M, 3 * DIM, DIM);

  // 2) RoPE + reshape/split
  int total = Bc * T * NHc * (HDc / 2);
  rope_split<<<(total + 255) / 256, 256, 0, stream>>>(qkv, sinp, cosp, qbf,
                                                      kbf, vtb, Bc, T, NHc, HDc);

  // 3) causal flash attention
  flash_attn<<<dim3(T / ATT_QTILE, Bc * NHc), 256, 0, stream>>>(qbf, kbf, vtb,
                                                                ybf, T);

  // 4) out = y @ W_proj   (4096 x 1024 x 1024)
  gemm_bf16_async<<<dim3(DIM / 128, M / 128), 256, 0, stream>>>(
      ybf, wprojT, out, M, DIM, DIM);
}
